// MultiHeadAttention_65446711656450
// MI455X (gfx1250) — matrix operand order repaired and checked by reading, not compiled, not run on hardware
//
#include <hip/hip_runtime.h>
#include <hip/hip_bf16.h>

// ---------------------------------------------------------------------------
// MI455X (gfx1250) multi-head attention, bf16 WMMA pipeline, f32 accumulate.
// B=4, S=1024, D=1024, H=16, E=64.
// GEMMs: double-buffered LDS with GLOBAL_LOAD_ASYNC_TO_LDS (ASYNCcnt path).
// ---------------------------------------------------------------------------

typedef __attribute__((ext_vector_type(16))) __bf16 v16bf;
typedef __attribute__((ext_vector_type(8)))  __bf16 v8bf;
typedef __attribute__((ext_vector_type(4)))  __bf16 v4bf;
typedef __attribute__((ext_vector_type(8)))  float  v8f;

#define Bn 4
#define Sn 1024
#define Dn 1024
#define Hn 16
#define En 64
#define Mn (Bn * Sn)         // 4096

// ---- WMMA wrapper: D = A(16x32 bf16) * B(32x16 bf16) + C(16x16 f32) -------
__device__ __forceinline__ v8f wmma_bf16(v16bf a, v16bf b, v8f c) {
  return __builtin_amdgcn_wmma_f32_16x16x32_bf16(
      /*neg_a=*/false, a, /*neg_b=*/false, b,
      /*c_mod=*/(short)0, c, /*reuse_a=*/false, /*reuse_b=*/false);
}

// ---- async global -> LDS 16B copy (CDNA5 GLOBAL_LOAD_ASYNC_TO_LDS_B128) ---
// VDST VGPR carries the LDS byte offset; low 32 bits of a flat shared-aperture
// address are exactly that offset.
__device__ __forceinline__ void async_cp16(const void* lds_dst, const void* gsrc) {
  unsigned int       loff = (unsigned int)(unsigned long long)lds_dst;
  unsigned long long ga   = (unsigned long long)gsrc;
  asm volatile("global_load_async_to_lds_b128 %0, %1, off"
               :: "v"(loff), "v"(ga)
               : "memory");
}

__device__ __forceinline__ void wait_async0() {
#if __has_builtin(__builtin_amdgcn_s_wait_asynccnt)
  __builtin_amdgcn_s_wait_asynccnt(0);
#else
  asm volatile("s_wait_asynccnt 0x0" ::: "memory");
#endif
}

// ---- A-matrix fragment (16x32 bf16), base = (row0, k0), row-major ---------
// lanes 0-15: M=lane, K in {0..7, 16..23}; lanes 16-31: M=lane-16, K in {8..15, 24..31}
__device__ __forceinline__ v16bf frag_a(const __bf16* base, int stride, int lane) {
  const int half = lane >> 4;
  const int r    = lane & 15;
  const __bf16* p = base + r * stride + half * 8;
  v8bf lo = *(const v8bf*)(p);
  v8bf hi = *(const v8bf*)(p + 16);
  v16bf f;
#pragma unroll
  for (int i = 0; i < 8; ++i) { f[i] = lo[i]; f[i + 8] = hi[i]; }
  return f;
}

// ---- B-matrix fragment (32x16 bf16) from row-major [N][K] storage ---------
// lane n (0-15) holds column n, K in {0..7, 16..23}; lanes 16-31 hold column n, K in {8..15, 24..31}
__device__ __forceinline__ v16bf frag_b(const __bf16* base, int stride, int lane) {
  const int half = lane >> 4;
  const int r    = lane & 15;
  const __bf16* p = base + r * stride + half * 8;
  v8bf lo = *(const v8bf*)(p);
  v8bf hi = *(const v8bf*)(p + 16);
  v16bf f;
#pragma unroll
  for (int i = 0; i < 8; ++i) { f[i] = lo[i]; f[i + 8] = hi[i]; }
  return f;
}

// ---------------------------------------------------------------------------
// f32 -> bf16 pack (vectorized, n multiple of 1024)
// ---------------------------------------------------------------------------
__global__ __launch_bounds__(256) void pack_bf16(const float* __restrict__ s,
                                                 __bf16* __restrict__ d, int n) {
  int i = (blockIdx.x * 256 + threadIdx.x) * 4;
  if (i + 3 >= n) return;
  float4 v = *(const float4*)(s + i);
  v4bf o;
  o[0] = (__bf16)v.x; o[1] = (__bf16)v.y; o[2] = (__bf16)v.z; o[3] = (__bf16)v.w;
  *(v4bf*)(d + i) = o;
}

// ---------------------------------------------------------------------------
// Tiled GEMM: C[m,n] = sum_k A16[m,k] * Bw[n,k]   (A row-major, Bw = weight rows)
// Block tile 128x128, 8 waves (4 M x 2 N), each wave 32x64 (2x4 WMMA tiles).
// Double-buffered LDS fed by async copies; one barrier per K-step.
// MODE 0: fused QKV store (Q,K row-major [B,H,S,E]; V transposed [B,H,E,S])
// MODE 1: f32 output + bias (final projection)
// ---------------------------------------------------------------------------
#define LDP 40   // padded LDS row stride (bf16 elems) to avoid bank conflicts
#define KSTEPS (Dn / 32)

template <int MODE>
__global__ __launch_bounds__(256) void gemm_bf16(
    const __bf16* __restrict__ A, const __bf16* __restrict__ Bw,
    __bf16* __restrict__ q16, __bf16* __restrict__ k16, __bf16* __restrict__ vt16,
    const float* __restrict__ bias, float* __restrict__ outf) {
  const int tid  = threadIdx.x;
  const int lane = tid & 31;
  const int w    = tid >> 5;            // wave 0..7
  const int wm   = w & 3;               // M position (4)
  const int wn   = w >> 2;              // N position (2)
  const int m0   = blockIdx.y * 128;
  const int n0   = blockIdx.x * 128;
  const int half = lane >> 4;
  const int l15  = lane & 15;

  __shared__ __bf16 As[2][128 * LDP];
  __shared__ __bf16 Bs[2][128 * LDP];

  // this thread's two staging chunks (row, k-offset) are loop-invariant
  const int c0 = tid, c1 = tid + 256;
  const int r0 = c0 >> 2, o0 = (c0 & 3) * 8;
  const int r1 = c1 >> 2, o1 = (c1 & 3) * 8;

  auto stage = [&](int k0, int buf) {
    async_cp16(&As[buf][r0 * LDP + o0], &A[(size_t)(m0 + r0) * Dn + k0 + o0]);
    async_cp16(&Bs[buf][r0 * LDP + o0], &Bw[(size_t)(n0 + r0) * Dn + k0 + o0]);
    async_cp16(&As[buf][r1 * LDP + o1], &A[(size_t)(m0 + r1) * Dn + k0 + o1]);
    async_cp16(&Bs[buf][r1 * LDP + o1], &Bw[(size_t)(n0 + r1) * Dn + k0 + o1]);
  };

  v8f acc[2][4] = {};

  stage(0, 0);
  for (int it = 0; it < KSTEPS; ++it) {
    wait_async0();       // this wave's portion of tile `it` has landed in LDS
    __syncthreads();     // every wave's portion visible; prev reads of buf^1 done
    if (it + 1 < KSTEPS) stage((it + 1) * 32, (it + 1) & 1);  // prefetch next

    const int buf = it & 1;
    v16bf af[2], bf[4];
#pragma unroll
    for (int mt = 0; mt < 2; ++mt)
      af[mt] = frag_a(&As[buf][(wm * 32 + mt * 16) * LDP], LDP, lane);
#pragma unroll
    for (int nt = 0; nt < 4; ++nt)
      bf[nt] = frag_b(&Bs[buf][(wn * 64 + nt * 16) * LDP], LDP, lane);

#pragma unroll
    for (int mt = 0; mt < 2; ++mt)
#pragma unroll
      for (int nt = 0; nt < 4; ++nt)
        acc[mt][nt] = wmma_bf16(af[mt], bf[nt], acc[mt][nt]);
  }

  // ---- store: C layout is M = i + 8*half, N = lane&15 per 16x16 tile ------
#pragma unroll
  for (int mt = 0; mt < 2; ++mt)
#pragma unroll
    for (int nt = 0; nt < 4; ++nt)
#pragma unroll
      for (int i = 0; i < 8; ++i) {
        int mrow = m0 + wm * 32 + mt * 16 + i + half * 8;
        int ncol = n0 + wn * 64 + nt * 16 + l15;
        float val = acc[mt][nt][i];
        if constexpr (MODE == 0) {
          int sec = ncol >> 10;          // 0=Q 1=K 2=V (uniform per block)
          int nn  = ncol & 1023;
          int h   = nn >> 6, e = nn & 63;
          int b   = mrow >> 10, s = mrow & 1023;
          if (sec == 0)
            q16[((size_t)(b * Hn + h) * Sn + s) * En + e] = (__bf16)val;
          else if (sec == 1)
            k16[((size_t)(b * Hn + h) * Sn + s) * En + e] = (__bf16)val;
          else
            vt16[((size_t)(b * Hn + h) * En + e) * Sn + s] = (__bf16)val;
        } else {
          outf[(size_t)mrow * Dn + ncol] = val + bias[ncol];
        }
      }
}

// ---------------------------------------------------------------------------
// Flash attention: grid.x = S/64 query tiles, grid.y = B*H.
// Block = 128 threads (4 waves); wave owns 16 query rows, 64-col key tiles.
// Causal: query tile i iterates i+1 key tiles only. K/V staged via async LDS.
// ---------------------------------------------------------------------------
#define KVP 72   // padded LDS stride for 64-wide tiles

__global__ __launch_bounds__(128) void flash_attn(
    const __bf16* __restrict__ Q, const __bf16* __restrict__ K,
    const __bf16* __restrict__ Vt, __bf16* __restrict__ Oo) {
  const int tid  = threadIdx.x;
  const int lane = tid & 31;
  const int w    = tid >> 5;             // wave 0..3
  const int bh   = blockIdx.y;           // b*H + h
  const int qb   = blockIdx.x * 64;
  const int qr0  = qb + w * 16;          // wave's first query row
  const int half = lane >> 4;
  const int l15  = lane & 15;

  __shared__ __bf16 Ks[64 * KVP];
  __shared__ __bf16 Vs[64 * KVP];
  __shared__ __bf16 Ps[4][16 * KVP];

  // Q fragments for E=64 (two K-steps of 32), loaded once, held in VGPRs
  const __bf16* qbase = Q + ((size_t)bh * Sn + qr0) * En;
  v16bf qf[2];
#pragma unroll
  for (int ks = 0; ks < 2; ++ks) qf[ks] = frag_a(qbase + ks * 32, En, lane);

  v8f o[4] = {};
  float mrun[8], lrun[8];
#pragma unroll
  for (int i = 0; i < 8; ++i) { mrun[i] = -__builtin_inff(); lrun[i] = 0.f; }

  const float sscale = 32.0f;            // sqrt(D) — faithful to reference bug
  const int nkt = blockIdx.x + 1;

  // staging chunks: 64 rows x 4 x 16B per tensor, 2 chunks/thread/tensor
  const int c0 = tid, c1 = tid + 128;
  const int r0 = c0 >> 2, o0 = (c0 & 3) * 8;
  const int r1 = c1 >> 2, o1 = (c1 & 3) * 8;

  for (int kt = 0; kt < nkt; ++kt) {
    const int t0 = kt * 64;
    __syncthreads();                     // previous tile's reads complete
    async_cp16(&Ks[r0 * KVP + o0], &K[((size_t)bh * Sn + t0 + r0) * En + o0]);
    async_cp16(&Vs[r0 * KVP + o0], &Vt[((size_t)bh * En + r0) * Sn + t0 + o0]);
    async_cp16(&Ks[r1 * KVP + o1], &K[((size_t)bh * Sn + t0 + r1) * En + o1]);
    async_cp16(&Vs[r1 * KVP + o1], &Vt[((size_t)bh * En + r1) * Sn + t0 + o1]);
    wait_async0();
    __syncthreads();                     // staged tiles visible to all waves

    // scores S = Q K^T  (B-matrix columns = key rows, contiguous in Ks)
    v8f sc[4] = {};
#pragma unroll
    for (int ks = 0; ks < 2; ++ks)
#pragma unroll
      for (int nt = 0; nt < 4; ++nt) {
        v16bf kf = frag_b(&Ks[(nt * 16) * KVP + ks * 32], KVP, lane);
        sc[nt] = wmma_bf16(qf[ks], kf, sc[nt]);
      }

    // scale + causal mask
#pragma unroll
    for (int nt = 0; nt < 4; ++nt)
#pragma unroll
      for (int i = 0; i < 8; ++i) {
        int col = t0 + nt * 16 + l15;
        int row = qr0 + i + half * 8;
        float v = sc[nt][i] * sscale;
        sc[nt][i] = (col <= row) ? v : -__builtin_inff();
      }

    // row max across the 16 lanes holding each row (half-wave reduction)
    float mx[8];
#pragma unroll
    for (int i = 0; i < 8; ++i) {
      float m = sc[0][i];
      m = fmaxf(m, sc[1][i]); m = fmaxf(m, sc[2][i]); m = fmaxf(m, sc[3][i]);
      mx[i] = m;
    }
#pragma unroll
    for (int off = 8; off >= 1; off >>= 1)
#pragma unroll
      for (int i = 0; i < 8; ++i)
        mx[i] = fmaxf(mx[i], __shfl_xor(mx[i], off, 16));

    float alpha[8], rsum[8];
#pragma unroll
    for (int i = 0; i < 8; ++i) {
      float mnew = fmaxf(mrun[i], mx[i]);
      alpha[i]   = __expf(mrun[i] - mnew);
      mrun[i]    = mnew;
      rsum[i]    = 0.f;
    }

    // P = exp(S - m), spill to LDS (C-layout -> A-layout conversion)
#pragma unroll
    for (int nt = 0; nt < 4; ++nt)
#pragma unroll
      for (int i = 0; i < 8; ++i) {
        float p = __expf(sc[nt][i] - mrun[i]);
        rsum[i] += p;
        Ps[w][(i + half * 8) * KVP + nt * 16 + l15] = (__bf16)p;
      }
#pragma unroll
    for (int off = 8; off >= 1; off >>= 1)
#pragma unroll
      for (int i = 0; i < 8; ++i)
        rsum[i] += __shfl_xor(rsum[i], off, 16);

#pragma unroll
    for (int i = 0; i < 8; ++i) lrun[i] = lrun[i] * alpha[i] + rsum[i];
#pragma unroll
    for (int nt = 0; nt < 4; ++nt)
#pragma unroll
      for (int i = 0; i < 8; ++i) o[nt][i] *= alpha[i];

    __syncthreads();   // P visible for fragment reads

    // O += P x V   (B-matrix columns = e, contiguous in Vs rows)
#pragma unroll
    for (int ks = 0; ks < 2; ++ks) {
      v16bf pf = frag_a(&Ps[w][ks * 32], KVP, lane);
#pragma unroll
      for (int nt = 0; nt < 4; ++nt) {
        v16bf vf = frag_b(&Vs[(nt * 16) * KVP + ks * 32], KVP, lane);
        o[nt] = wmma_bf16(pf, vf, o[nt]);
      }
    }
  }

  // normalize and store concat-head layout [B,S,D] in bf16
  const int b = bh >> 4, h = bh & 15;
#pragma unroll
  for (int nt = 0; nt < 4; ++nt)
#pragma unroll
    for (int i = 0; i < 8; ++i) {
      int s = qr0 + i + half * 8;
      int e = nt * 16 + l15;
      float val = o[nt][i] / lrun[i];
      Oo[((size_t)b * Sn + s) * Dn + h * En + e] = (__bf16)val;
    }
}

// ---------------------------------------------------------------------------
// launch
// ---------------------------------------------------------------------------
extern "C" void kernel_launch(void* const* d_in, const int* in_sizes, int n_in,
                              void* d_out, int out_size, void* d_ws, size_t ws_size,
                              hipStream_t stream) {
  const float* x  = (const float*)d_in[0];
  const float* Wq = (const float*)d_in[1];
  const float* Wk = (const float*)d_in[2];
  const float* Wv = (const float*)d_in[3];
  const float* Wp = (const float*)d_in[4];
  const float* bp = (const float*)d_in[5];

  char* ws = (char*)d_ws;
  const size_t MB = 1u << 20;
  __bf16* X16   = (__bf16*)(ws + 0 * MB);    // 8 MB  [4096,1024]
  __bf16* Wqkv  = (__bf16*)(ws + 8 * MB);    // 6 MB  [3072,1024] rows = out
  __bf16* Wp16  = (__bf16*)(ws + 14 * MB);   // 2 MB  [1024,1024]
  __bf16* Q16   = (__bf16*)(ws + 16 * MB);   // 8 MB  [B,H,S,E]
  __bf16* K16   = (__bf16*)(ws + 24 * MB);   // 8 MB  [B,H,S,E]
  __bf16* Vt16  = (__bf16*)(ws + 32 * MB);   // 8 MB  [B,H,E,S]
  __bf16* A16   = (__bf16*)(ws + 40 * MB);   // 8 MB  [B,S,D]

  auto pack = [&](const float* s, __bf16* d, int n) {
    pack_bf16<<<n / 1024, 256, 0, stream>>>(s, d, n);
  };
  pack(x,  X16,                    Bn * Sn * Dn);
  pack(Wq, Wqkv,                   Hn * En * Dn);
  pack(Wk, Wqkv + 1024 * 1024,     Hn * En * Dn);
  pack(Wv, Wqkv + 2 * 1024 * 1024, Hn * En * Dn);
  pack(Wp, Wp16,                   Dn * Dn);

  // fused QKV projection: [4096 x 1024] x [3072 x 1024]^T
  gemm_bf16<0><<<dim3(3072 / 128, Mn / 128), 256, 0, stream>>>(
      X16, Wqkv, Q16, K16, Vt16, nullptr, nullptr);

  // flash attention: 16 query tiles x 64 (b,h)
  flash_attn<<<dim3(Sn / 64, Bn * Hn), 128, 0, stream>>>(Q16, K16, Vt16, A16);

  // output projection + bias -> f32
  gemm_bf16<1><<<dim3(Dn / 128, Mn / 128), 256, 0, stream>>>(
      A16, Wp16, nullptr, nullptr, nullptr, bp, (float*)d_out);
}
